// _SelectiveSSM_37452114821301
// MI455X (gfx1250) — compile-verified, matrix operand order checked by reading, not run on hardware
//
#include <hip/hip_runtime.h>
#include <hip/hip_bf16.h>

typedef __attribute__((ext_vector_type(16))) __bf16 v16bf;
typedef __attribute__((ext_vector_type(8)))  float  v8f;
typedef int i32x4 __attribute__((vector_size(16)));

#if __has_builtin(__builtin_amdgcn_global_load_async_to_lds_b128) && \
    __has_builtin(__builtin_amdgcn_s_wait_asynccnt)
#define USE_ASYNC_LDS 1
#endif

__device__ __forceinline__ unsigned short f2bf(float f) {
    unsigned int u = __float_as_uint(f);
    unsigned int r = u + 0x7FFFu + ((u >> 16) & 1u);   // round-to-nearest-even
    return (unsigned short)(r >> 16);
}

// 16-byte global -> LDS copy (async engine if available)
__device__ __forceinline__ void cp16_g2l(const void* g, void* l) {
#ifdef USE_ASYNC_LDS
    __builtin_amdgcn_global_load_async_to_lds_b128(
        (__attribute__((address_space(1))) i32x4*)g,
        (__attribute__((address_space(3))) i32x4*)l, 0, 0);
#else
    *(uint4*)l = *(const uint4*)g;
#endif
}

__device__ __forceinline__ void async_wait_all() {
#ifdef USE_ASYNC_LDS
    __builtin_amdgcn_s_wait_asynccnt(0);
#endif
}

// ---------------------------------------------------------------- fp32 -> bf16
__global__ __launch_bounds__(256) void k_to_bf16(const float* __restrict__ src,
                                                 unsigned short* __restrict__ dst,
                                                 size_t n) {
    size_t i = (size_t)blockIdx.x * blockDim.x + threadIdx.x;
    if (i < n) dst[i] = f2bf(src[i]);
}

// ---------------------------------------------------------------- WMMA GEMM
// C[M,N] = A[M,K] * W[N,K]^T   (A,W bf16 row-major, accum f32)
// Block: 256 threads = 8 waves. Block tile 128(M) x 64(N), K-stage 64,
// double-buffered LDS filled by the async global->LDS engine.
// Each wave: one 16-row strip, 4 wmma tiles along N (A fragment reused).
// All fragments for a K-stage are loaded from LDS into registers first
// (single coarse DS wait), then 8 WMMAs issue as an uninterrupted burst.
// mode 0: split columns into out0 (n<di) / out1 (n>=di)
// mode 1: out0 = softplus(acc + bias[n])
// mode 2: out0 = acc
__global__ __launch_bounds__(256) void k_gemm_bf16_wmma(
    const unsigned short* __restrict__ A,
    const unsigned short* __restrict__ W,
    int M, int N, int K, int mode,
    const float* __restrict__ bias,
    float* __restrict__ out0, float* __restrict__ out1, int di) {

    __shared__ __align__(16) unsigned short lA[2][128 * 64];
    __shared__ __align__(16) unsigned short lB[2][64 * 64];

    const int tid  = threadIdx.x;
    const int lane = tid & 31;
    const int wave = tid >> 5;
    const int half = lane >> 4;
    const int l15  = lane & 15;

    const int mBlock = blockIdx.y * 128;
    const int nBlock = blockIdx.x * 64;

    v8f acc0 = {}; v8f acc1 = {}; v8f acc2 = {}; v8f acc3 = {};

    const int nStages = K >> 6;

    // ---- stage copy: A tile 128x64 (4 x b128 per thread), B tile 64x64 (2 x b128)
    auto stage_copy = [&](int s, int buf) {
        int kt = s << 6;
#pragma unroll
        for (int t = 0; t < 4; ++t) {
            int linear = t * 256 + tid;        // 0..1023
            int r = linear >> 3, c = linear & 7;
            cp16_g2l(A + (size_t)(mBlock + r) * K + kt + c * 8,
                     &lA[buf][r * 64 + c * 8]);
        }
#pragma unroll
        for (int t = 0; t < 2; ++t) {
            int linear = t * 256 + tid;        // 0..511
            int n = linear >> 3, c = linear & 7;
            cp16_g2l(W + (size_t)(nBlock + n) * K + kt + c * 8,
                     &lB[buf][n * 64 + c * 8]);
        }
    };

    stage_copy(0, 0);
    async_wait_all();
    __syncthreads();

    union Frag { uint4 q[2]; v16bf v; };

    for (int s = 0; s < nStages; ++s) {
        const int buf = s & 1;
        if (s + 1 < nStages) stage_copy(s + 1, buf ^ 1);

        // ---- load ALL fragments for this 64-K stage into registers
        const int arow = wave * 16 + l15;
        Frag a[2];
        Frag b[2][4];
#pragma unroll
        for (int kh = 0; kh < 2; ++kh) {        // kh: 32-k step index
            const int kk = kh * 32;
            a[kh].q[0] = *(const uint4*)&lA[buf][arow * 64 + kk + half * 8];
            a[kh].q[1] = *(const uint4*)&lA[buf][arow * 64 + kk + 16 + half * 8];
#pragma unroll
            for (int nt = 0; nt < 4; ++nt) {
                const int bcol = nt * 16 + l15;
                b[kh][nt].q[0] = *(const uint4*)&lB[buf][bcol * 64 + kk + half * 8];
                b[kh][nt].q[1] = *(const uint4*)&lB[buf][bcol * 64 + kk + 16 + half * 8];
            }
        }

        // ---- 8 back-to-back WMMAs (independent accumulators)
#pragma unroll
        for (int kh = 0; kh < 2; ++kh) {
            acc0 = __builtin_amdgcn_wmma_f32_16x16x32_bf16(false, a[kh].v, false, b[kh][0].v,
                                                           (short)0, acc0, false, false);
            acc1 = __builtin_amdgcn_wmma_f32_16x16x32_bf16(false, a[kh].v, false, b[kh][1].v,
                                                           (short)0, acc1, false, false);
            acc2 = __builtin_amdgcn_wmma_f32_16x16x32_bf16(false, a[kh].v, false, b[kh][2].v,
                                                           (short)0, acc2, false, false);
            acc3 = __builtin_amdgcn_wmma_f32_16x16x32_bf16(false, a[kh].v, false, b[kh][3].v,
                                                           (short)0, acc3, false, false);
        }

        async_wait_all();
        __syncthreads();
    }

    // Epilogue. D layout: vgpr r -> m = r + 8*half, n = lane&15.
#pragma unroll
    for (int nt = 0; nt < 4; ++nt) {
        v8f c = (nt == 0) ? acc0 : (nt == 1) ? acc1 : (nt == 2) ? acc2 : acc3;
#pragma unroll
        for (int r = 0; r < 8; ++r) {
            int m = mBlock + wave * 16 + r + 8 * half;
            int n = nBlock + nt * 16 + l15;
            float v = c[r];
            if (mode == 0) {
                if (n < di) out0[(size_t)m * di + n]        = v;
                else        out1[(size_t)m * di + (n - di)] = v;
            } else if (mode == 1) {
                float x = v + bias[n];
                float sp = (x > 20.f) ? x : __logf(1.f + __expf(x));
                out0[(size_t)m * N + n] = sp;
            } else {
                out0[(size_t)m * N + n] = v;
            }
        }
    }
}

// ---------------------------------------------------------------- depthwise causal conv + silu
__global__ __launch_bounds__(256) void k_conv_silu(
    const float* __restrict__ xp, const float* __restrict__ wc,
    float* __restrict__ xc, unsigned short* __restrict__ xcbf,
    int L, int di, size_t total) {
    size_t idx = (size_t)blockIdx.x * blockDim.x + threadIdx.x;
    if (idx >= total) return;
    int d = (int)(idx % di);
    int t = (int)((idx / di) % L);
    size_t bBase = (idx / ((size_t)di * L)) * (size_t)L * di;
    float s = 0.f;
#pragma unroll
    for (int j = 0; j < 4; ++j) {
        int tt = t - 3 + j;
        if (tt >= 0) s += xp[bBase + (size_t)tt * di + d] * wc[d * 4 + j];
    }
    float v = s / (1.f + __expf(-s));      // silu
    xc[idx]   = v;
    xcbf[idx] = f2bf(v);
}

// ---------------------------------------------------------------- x_dbl = xc @ W_x^T  (N = 32)
__global__ __launch_bounds__(256) void k_xdbl(
    const float* __restrict__ xc, const float* __restrict__ Wx,
    float* __restrict__ xdbl, int di) {
    __shared__ float sx[8][128];
    int tid  = threadIdx.x;
    int rloc = tid >> 5;
    int j    = tid & 31;
    size_t row = (size_t)blockIdx.x * 8 + rloc;
    float acc = 0.f;
    for (int k0 = 0; k0 < di; k0 += 128) {
        __syncthreads();
        for (int i = tid; i < 8 * 128; i += 256) {
            int rr = i >> 7, kk = i & 127;
            sx[rr][kk] = xc[((size_t)blockIdx.x * 8 + rr) * di + k0 + kk];
        }
        __syncthreads();
#pragma unroll 8
        for (int kk = 0; kk < 128; ++kk)
            acc += sx[rloc][kk] * Wx[(size_t)j * di + k0 + kk];
    }
    xdbl[row * 32 + j] = acc;
}

// ---------------------------------------------------------------- selective scan (sequential in L)
__global__ __launch_bounds__(256) void k_scan(
    const float* __restrict__ delta, const float* __restrict__ xdbl,
    const float* __restrict__ z, const float* __restrict__ xc,
    const float* __restrict__ A_log, const float* __restrict__ Dp,
    unsigned short* __restrict__ ybf, int L, int di) {

    int b = blockIdx.y;
    int d = blockIdx.x * 256 + threadIdx.x;

    float Acoef[16], h[16];
#pragma unroll
    for (int s = 0; s < 16; ++s) { Acoef[s] = -__expf(A_log[s]); h[s] = 0.f; }
    float Dd = Dp[d];

    __shared__ float sB[64][16];
    __shared__ float sC[64][16];
    const size_t base = (size_t)b * L;

    for (int t0 = 0; t0 < L; t0 += 64) {
        __syncthreads();
        for (int i = threadIdx.x; i < 64 * 32; i += 256) {
            int tt = i >> 5, jj = i & 31;
            float v = xdbl[(base + t0 + tt) * 32 + jj];
            if (jj < 16) sB[tt][jj]      = v;
            else         sC[tt][jj - 16] = v;
        }
        __syncthreads();
        for (int tl = 0; tl < 64; ++tl) {
            size_t idx = (base + t0 + tl) * di + d;
            float de  = delta[idx];
            float acc = 0.f;
#pragma unroll
            for (int s = 0; s < 16; ++s) {
                float dA = __expf(de * Acoef[s]);
                h[s] = dA * h[s] + de * sB[tl][s];
                acc += h[s] * sC[tl][s];
            }
            float zz = z[idx];
            float g  = zz / (1.f + __expf(-zz));           // silu(z)
            float y  = (acc + Dd * xc[idx]) * g;
            ybf[idx] = f2bf(y);
        }
    }
}

// ---------------------------------------------------------------- launcher
extern "C" void kernel_launch(void* const* d_in, const int* in_sizes, int n_in,
                              void* d_out, int out_size, void* d_ws, size_t ws_size,
                              hipStream_t stream) {
    (void)in_sizes; (void)n_in; (void)out_size; (void)ws_size;

    const int B = 2, L = 2048, Dm = 768, di = 1536;
    const int M = B * L;              // 4096
    const int N1 = 2 * di;            // 3072

    const float* x     = (const float*)d_in[0];
    const float* W_in  = (const float*)d_in[1];
    const float* W_cv  = (const float*)d_in[2];
    const float* W_x   = (const float*)d_in[3];
    const float* W_dt  = (const float*)d_in[4];
    const float* b_dt  = (const float*)d_in[5];
    const float* A_log = (const float*)d_in[6];
    const float* Dp    = (const float*)d_in[7];
    const float* W_out = (const float*)d_in[8];
    float* out = (float*)d_out;

    // workspace carve-out (256B aligned)
    char* ws = (char*)d_ws;
    size_t off = 0;
    auto alloc = [&](size_t bytes) { size_t o = off; off += (bytes + 255) & ~(size_t)255; return o; };

    unsigned short* xbf    = (unsigned short*)(ws + alloc((size_t)M * Dm * 2));
    unsigned short* wInBf  = (unsigned short*)(ws + alloc((size_t)N1 * Dm * 2));
    unsigned short* wDtBf  = (unsigned short*)(ws + alloc((size_t)di * di * 2));
    unsigned short* wOutBf = (unsigned short*)(ws + alloc((size_t)Dm * di * 2));
    float* xpart = (float*)(ws + alloc((size_t)M * di * 4));   // reused as delta
    float* zbuf  = (float*)(ws + alloc((size_t)M * di * 4));
    float* xc    = (float*)(ws + alloc((size_t)M * di * 4));
    unsigned short* xcbf = (unsigned short*)(ws + alloc((size_t)M * di * 2));
    float* xdbl  = (float*)(ws + alloc((size_t)M * 32 * 4));
    unsigned short* ybf = (unsigned short*)(ws + alloc((size_t)M * di * 2));
    float* delta = xpart;   // xpart dead after conv

    // 1) pack to bf16
    {
        size_t n;
        n = (size_t)M * Dm;   k_to_bf16<<<(unsigned)((n + 255) / 256), 256, 0, stream>>>(x,     xbf,    n);
        n = (size_t)N1 * Dm;  k_to_bf16<<<(unsigned)((n + 255) / 256), 256, 0, stream>>>(W_in,  wInBf,  n);
        n = (size_t)di * di;  k_to_bf16<<<(unsigned)((n + 255) / 256), 256, 0, stream>>>(W_dt,  wDtBf,  n);
        n = (size_t)Dm * di;  k_to_bf16<<<(unsigned)((n + 255) / 256), 256, 0, stream>>>(W_out, wOutBf, n);
    }

    // 2) xz = x @ W_in^T, split into x_part / z
    k_gemm_bf16_wmma<<<dim3(N1 / 64, M / 128), 256, 0, stream>>>(
        xbf, wInBf, M, N1, Dm, /*mode=*/0, nullptr, xpart, zbuf, di);

    // 3) depthwise causal conv + silu
    {
        size_t total = (size_t)M * di;
        k_conv_silu<<<(unsigned)((total + 255) / 256), 256, 0, stream>>>(
            xpart, W_cv, xc, xcbf, L, di, total);
    }

    // 4) delta = softplus(xc @ W_dt^T + b_dt)   (overwrites xpart buffer)
    k_gemm_bf16_wmma<<<dim3(di / 64, M / 128), 256, 0, stream>>>(
        xcbf, wDtBf, M, di, di, /*mode=*/1, b_dt, delta, nullptr, di);

    // 5) x_dbl = xc @ W_x^T  (B_ssm | C_ssm)
    k_xdbl<<<M / 8, 256, 0, stream>>>(xc, W_x, xdbl, di);

    // 6) selective scan + D-skip + silu(z) gate -> bf16
    k_scan<<<dim3(di / 256, B), 256, 0, stream>>>(
        delta, xdbl, zbuf, xc, A_log, Dp, ybf, L, di);

    // 7) out = y @ W_out^T
    k_gemm_bf16_wmma<<<dim3(Dm / 64, M / 128), 256, 0, stream>>>(
        ybf, wOutBf, M, Dm, di, /*mode=*/2, nullptr, out, nullptr, di);
}
